// MultiHeadSelfAttention_6425271075015
// MI455X (gfx1250) — compile-verified, matrix operand order checked by reading, not run on hardware
//
#include <hip/hip_runtime.h>

#define BB 4
#define TT 1024
#define CC 1024
#define HH 16
#define HD 64

// padded LDS row stride (shorts) for the staged K slab: 64 data + 8 pad
#define KROW 72

typedef __attribute__((ext_vector_type(16))) __bf16 v16bf;
typedef __attribute__((ext_vector_type(8)))  __bf16 v8bf;
typedef __attribute__((ext_vector_type(16))) float v16f;
typedef __attribute__((ext_vector_type(8)))  float v8f;

static __device__ __forceinline__ unsigned short f2bf(float f) {
    unsigned int u = __builtin_bit_cast(unsigned int, f);
    unsigned int r = u + 0x7FFFu + ((u >> 16) & 1u);   // round-to-nearest-even
    return (unsigned short)(r >> 16);
}

static __device__ __forceinline__ v16bf ldfrag(const unsigned short* p) {
    return *reinterpret_cast<const v16bf*>(p);
}

// 16B-aligned fragment load (for padded LDS rows)
static __device__ __forceinline__ v16bf ldfrag16(const unsigned short* p) {
    union { v16bf v; v8bf h[2]; } u;
    u.h[0] = *reinterpret_cast<const v8bf*>(p);
    u.h[1] = *reinterpret_cast<const v8bf*>(p + 8);
    return u.v;
}

// load 16 f32 and convert to a bf16 fragment
static __device__ __forceinline__ v16bf cvtfrag(const float* p) {
    v16f f = *reinterpret_cast<const v16f*>(p);
    union { v16bf bf; unsigned short us[16]; } u;
#pragma unroll
    for (int i = 0; i < 16; ++i) u.us[i] = f2bf(f[i]);
    return u.bf;
}

static __device__ __forceinline__ v8f wmma_bf16(v16bf a, v16bf b, v8f c) {
    return __builtin_amdgcn_wmma_f32_16x16x32_bf16(false, a, false, b, (short)0, c, false, false);
}

// ---------------- conversion kernels ----------------

__global__ void k_cvt_x(const float* __restrict__ x, unsigned short* __restrict__ o, int n) {
    for (int i = blockIdx.x * blockDim.x + threadIdx.x; i < n; i += gridDim.x * blockDim.x)
        o[i] = f2bf(x[i]);
}

// W[p][h][c][d] (each [H,C,HD]) -> wt[p][h][d][c]  (bf16, transposed for contiguous B-column loads)
__global__ void k_cvt_w(const float* __restrict__ wq, const float* __restrict__ wk,
                        const float* __restrict__ wv, unsigned short* __restrict__ o) {
    const int n = 3 * HH * HD * CC;
    for (int i = blockIdx.x * blockDim.x + threadIdx.x; i < n; i += gridDim.x * blockDim.x) {
        int c = i & (CC - 1);
        int d = (i >> 10) & (HD - 1);
        int h = (i >> 16) & (HH - 1);
        int p = i >> 20;
        const float* src = (p == 0) ? wq : ((p == 1) ? wk : wv);
        o[i] = f2bf(src[((size_t)h * CC + c) * HD + d]);
    }
}

// W_proj[k][n] -> wpt[n][k]
__global__ void k_cvt_wp(const float* __restrict__ wp, unsigned short* __restrict__ o) {
    const int n = CC * CC;
    for (int i = blockIdx.x * blockDim.x + threadIdx.x; i < n; i += gridDim.x * blockDim.x) {
        int k = i & (CC - 1);
        int nn = i >> 10;
        o[i] = f2bf(wp[(size_t)k * CC + nn]);
    }
}

// ---------------- QKV projection: q,k:[B,H,T,HD], vT:[B,H,HD,T] (bf16) ----------------
// one wave per 16x64 output block (4 WMMA tiles share one A fragment)
__global__ void k_qkv(const unsigned short* __restrict__ xb, const unsigned short* __restrict__ wt,
                      unsigned short* __restrict__ qb, unsigned short* __restrict__ kb,
                      unsigned short* __restrict__ vtb) {
    const int lane = threadIdx.x & 31;
    const int wid  = blockIdx.x * 4 + (threadIdx.x >> 5);  // 12288 waves total
    const int tt = wid & 63, h = (wid >> 6) & 15, b = (wid >> 10) & 3, p = wid >> 12;
    const int t0 = tt * 16;
    const int hi = lane >> 4, mn = lane & 15;

    const unsigned short* arow = xb + ((size_t)(b * TT + t0 + mn)) * CC + 16 * hi;
    const unsigned short* brow[4];
#pragma unroll
    for (int nt = 0; nt < 4; ++nt)
        brow[nt] = wt + ((size_t)((p * HH + h) * HD + nt * 16 + mn)) * CC + 16 * hi;

    v8f acc[4] = {};
#pragma unroll 2
    for (int kk = 0; kk < CC; kk += 32) {
        v16bf a = ldfrag(arow + kk);
#pragma unroll
        for (int nt = 0; nt < 4; ++nt)
            acc[nt] = wmma_bf16(a, ldfrag(brow[nt] + kk), acc[nt]);
    }

    if (p == 0) {  // q, fold 1/sqrt(HD)
#pragma unroll
        for (int nt = 0; nt < 4; ++nt)
#pragma unroll
            for (int r = 0; r < 8; ++r)
                qb[((size_t)((b * HH + h) * TT + t0 + r + 8 * hi)) * HD + nt * 16 + mn] =
                    f2bf(acc[nt][r] * 0.125f);
    } else if (p == 1) {
#pragma unroll
        for (int nt = 0; nt < 4; ++nt)
#pragma unroll
            for (int r = 0; r < 8; ++r)
                kb[((size_t)((b * HH + h) * TT + t0 + r + 8 * hi)) * HD + nt * 16 + mn] =
                    f2bf(acc[nt][r]);
    } else {  // v stored transposed [B,H,HD,T]
#pragma unroll
        for (int nt = 0; nt < 4; ++nt)
#pragma unroll
            for (int r = 0; r < 8; ++r)
                vtb[((size_t)((b * HH + h) * HD + nt * 16 + mn)) * TT + t0 + r + 8 * hi] =
                    f2bf(acc[nt][r]);
    }
}

// ---------------- fused scores + mask + softmax: writes normalized att once ----------------
// one workgroup (8 waves) per (b,h,16-row block); wave wv owns s-columns [wv*128, wv*128+128)
// K slab for this (b,h) is staged in LDS via async global->LDS DMA (ASYNCcnt path)
__global__ void k_attn(const unsigned short* __restrict__ qb, const unsigned short* __restrict__ kb,
                       const unsigned char* __restrict__ mask, const int* __restrict__ causal,
                       float* __restrict__ att) {
    extern __shared__ unsigned short ksh[];                  // [TT][KROW] + reductions
    float* rmax = (float*)(ksh + TT * KROW);
    float* rsum = rmax + 128;

    const int tid  = threadIdx.x;
    const int lane = tid & 31;
    const int wv   = tid >> 5;
    const int wid  = blockIdx.x;                             // 4096 blocks
    const int tt = wid & 63, h = (wid >> 6) & 15, b = wid >> 10;
    const int t0 = tt * 16;
    const int hi = lane >> 4, mn = lane & 15;
    const int cz = causal[0];
    const float NINF = -__builtin_inff();
    const size_t bh = (size_t)(b * HH + h);

    // ---- stage k rows [0, rows_needed) into LDS with async 16B copies ----
    const unsigned short* ksrc = kb + bh * TT * HD;
    const int rows_needed = cz ? (t0 + 16) : TT;             // causal: only s <= t0+15 used
    const int chunks = rows_needed * 8;                      // 8 x 16B per row
    const unsigned int kbase = (unsigned int)(uintptr_t)ksh; // workgroup-relative LDS byte offset
    const int iters = (chunks + 255) >> 8;
    for (int it = 0; it < iters; ++it) {
        const int c = (it << 8) + tid;
        if (c < chunks) {
            const int row = c >> 3, part = c & 7;
            const unsigned int ldsoff = kbase + (unsigned int)(row * (KROW * 2) + part * 16);
            const unsigned int goff   = (unsigned int)(row * (HD * 2) + part * 16);
            asm volatile("global_load_async_to_lds_b128 %0, %1, %2"
                         :: "v"(ldsoff), "v"(goff), "s"(ksrc) : "memory");
        }
    }
    asm volatile("s_wait_asynccnt 0x0" ::: "memory");
    __syncthreads();

    const unsigned short* qrow = qb + (bh * TT + t0 + mn) * HD + 16 * hi;
    const v16bf qa0 = ldfrag(qrow);
    const v16bf qa1 = ldfrag(qrow + 32);

    v8f accs[8];
#pragma unroll
    for (int i = 0; i < 8; ++i) {
        const int s0 = (wv * 8 + i) * 16;
        if (cz && s0 > t0 + 15) {                    // wave-uniform: fully masked tile
            v8f t;
#pragma unroll
            for (int r = 0; r < 8; ++r) t[r] = NINF;
            accs[i] = t;
        } else {
            const unsigned short* krow = ksh + (s0 + mn) * KROW + 16 * hi;
            v8f acc = {};
            acc = wmma_bf16(qa0, ldfrag16(krow), acc);
            acc = wmma_bf16(qa1, ldfrag16(krow + 32), acc);
            const int s = s0 + mn;
            const bool mok = mask[b * TT + s] != 0;
#pragma unroll
            for (int r = 0; r < 8; ++r) {
                const int t = t0 + r + 8 * hi;
                if (!(mok && (!cz || s <= t))) acc[r] = NINF;
            }
            accs[i] = acc;
        }
    }

    // per-row max over this wave's 128 columns (reduce 8 tiles, then 16 lanes of the half)
    float pm[8];
#pragma unroll
    for (int r = 0; r < 8; ++r) {
        float m = accs[0][r];
#pragma unroll
        for (int i = 1; i < 8; ++i) m = fmaxf(m, accs[i][r]);
#pragma unroll
        for (int off = 8; off >= 1; off >>= 1) m = fmaxf(m, __shfl_xor(m, off, 32));
        pm[r] = m;
    }
    if (mn == 0) {
#pragma unroll
        for (int r = 0; r < 8; ++r) rmax[wv * 16 + r + 8 * hi] = pm[r];
    }
    __syncthreads();

    float M[8], ps[8];
#pragma unroll
    for (int r = 0; r < 8; ++r) {
        float m = rmax[r + 8 * hi];
#pragma unroll
        for (int w = 1; w < 8; ++w) m = fmaxf(m, rmax[w * 16 + r + 8 * hi]);
        M[r] = m;
        ps[r] = 0.f;
    }

#pragma unroll
    for (int i = 0; i < 8; ++i)
#pragma unroll
        for (int r = 0; r < 8; ++r) {
            float e = __expf(accs[i][r] - M[r]);     // exp(-inf - finite) == 0
            accs[i][r] = e;
            ps[r] += e;
        }

#pragma unroll
    for (int r = 0; r < 8; ++r) {
        float s = ps[r];
#pragma unroll
        for (int off = 8; off >= 1; off >>= 1) s += __shfl_xor(s, off, 32);
        ps[r] = s;
    }
    if (mn == 0) {
#pragma unroll
        for (int r = 0; r < 8; ++r) rsum[wv * 16 + r + 8 * hi] = ps[r];
    }
    __syncthreads();

    float inv[8];
#pragma unroll
    for (int r = 0; r < 8; ++r) {
        float s = 0.f;
#pragma unroll
        for (int w = 0; w < 8; ++w) s += rsum[w * 16 + r + 8 * hi];
        inv[r] = 1.0f / s;
    }

    float* arow = att + (bh * TT) * TT;
#pragma unroll
    for (int i = 0; i < 8; ++i) {
        const int s0 = (wv * 8 + i) * 16;
#pragma unroll
        for (int r = 0; r < 8; ++r)
            arow[(size_t)(t0 + r + 8 * hi) * TT + s0 + mn] = accs[i][r] * inv[r];
    }
}

// ---------------- att @ v : hout[b,t, h*HD+d] (bf16) ----------------
// one wave per 16x64 block: the converted att A-fragment is reused by 4 WMMAs
__global__ void k_av(const float* __restrict__ att, const unsigned short* __restrict__ vtb,
                     const int* __restrict__ causal, unsigned short* __restrict__ houtb) {
    const int lane = threadIdx.x & 31;
    const int wid  = blockIdx.x * 8 + (threadIdx.x >> 5);  // 4096 waves
    const int tt = wid & 63, h = (wid >> 6) & 15, b = wid >> 10;
    const int t0 = tt * 16;
    const int hi = lane >> 4, mn = lane & 15;
    const int cz = causal[0];
    const int kend = cz ? ((t0 + 16 + 31) & ~31) : TT;  // masked cols are exactly 0 after softmax
    const size_t bh = (size_t)(b * HH + h);

    const float* arow = att + (bh * TT + t0 + mn) * TT + 16 * hi;
    const unsigned short* brow[4];
#pragma unroll
    for (int nt = 0; nt < 4; ++nt)
        brow[nt] = vtb + (bh * HD + nt * 16 + mn) * TT + 16 * hi;

    v8f acc[4] = {};
    for (int kk = 0; kk < kend; kk += 32) {
        v16bf a = cvtfrag(arow + kk);
#pragma unroll
        for (int nt = 0; nt < 4; ++nt)
            acc[nt] = wmma_bf16(a, ldfrag(brow[nt] + kk), acc[nt]);
    }

#pragma unroll
    for (int nt = 0; nt < 4; ++nt)
#pragma unroll
        for (int r = 0; r < 8; ++r)
            houtb[((size_t)(b * TT + t0 + r + 8 * hi)) * CC + h * HD + nt * 16 + mn] =
                f2bf(acc[nt][r]);
}

// ---------------- out = hout @ W_proj + b_proj ----------------
// one wave per 16x64 block
__global__ void k_proj(const unsigned short* __restrict__ houtb, const unsigned short* __restrict__ wpt,
                       const float* __restrict__ bias, float* __restrict__ out) {
    const int lane = threadIdx.x & 31;
    const int wid  = blockIdx.x * 8 + (threadIdx.x >> 5);  // 4096 waves
    const int ng = wid & 15, rt = wid >> 4;                // rt over B*T/16 = 256
    const int row0 = rt * 16, n0 = ng * 64;
    const int hi = lane >> 4, mn = lane & 15;

    const unsigned short* arow = houtb + ((size_t)(row0 + mn)) * CC + 16 * hi;
    const unsigned short* brow[4];
#pragma unroll
    for (int nt = 0; nt < 4; ++nt)
        brow[nt] = wpt + ((size_t)(n0 + nt * 16 + mn)) * CC + 16 * hi;

    v8f acc[4] = {};
#pragma unroll 2
    for (int kk = 0; kk < CC; kk += 32) {
        v16bf a = ldfrag(arow + kk);
#pragma unroll
        for (int nt = 0; nt < 4; ++nt)
            acc[nt] = wmma_bf16(a, ldfrag(brow[nt] + kk), acc[nt]);
    }

#pragma unroll
    for (int nt = 0; nt < 4; ++nt) {
        const float bv = bias[n0 + nt * 16 + mn];
#pragma unroll
        for (int r = 0; r < 8; ++r)
            out[((size_t)(row0 + r + 8 * hi)) * CC + n0 + nt * 16 + mn] = acc[nt][r] + bv;
    }
}

extern "C" void kernel_launch(void* const* d_in, const int* in_sizes, int n_in,
                              void* d_out, int out_size, void* d_ws, size_t ws_size,
                              hipStream_t stream) {
    const float* x       = (const float*)d_in[0];
    const unsigned char* mask = (const unsigned char*)d_in[1];
    const float* Wq      = (const float*)d_in[2];
    const float* Wk      = (const float*)d_in[3];
    const float* Wv      = (const float*)d_in[4];
    const float* Wp      = (const float*)d_in[5];
    const float* bproj   = (const float*)d_in[6];
    const int*   causal  = (const int*)d_in[7];

    float* out = (float*)d_out;                       // [B,T,C]
    float* att = out + (size_t)BB * TT * CC;          // [B,H,T,T]

    // workspace layout (bf16 = unsigned short)
    char* wsb = (char*)d_ws;
    unsigned short* xb    = (unsigned short*)(wsb);                       //  8 MB  x
    unsigned short* wqkvt = (unsigned short*)(wsb + (8u   << 20));        //  6 MB  W qkv^T
    unsigned short* wpt   = (unsigned short*)(wsb + (14u  << 20));        //  2 MB  W_proj^T
    unsigned short* qb    = (unsigned short*)(wsb + (16u  << 20));        //  8 MB  q*0.125
    unsigned short* kb    = (unsigned short*)(wsb + (24u  << 20));        //  8 MB  k
    unsigned short* vtb   = (unsigned short*)(wsb + (32u  << 20));        //  8 MB  v^T
    unsigned short* houtb = (unsigned short*)(wsb + (40u  << 20));        //  8 MB  concat heads

    // conversions
    k_cvt_x <<<2048, 256, 0, stream>>>(x, xb, BB * TT * CC);
    k_cvt_w <<<2048, 256, 0, stream>>>(Wq, Wk, Wv, wqkvt);
    k_cvt_wp<<<1024, 256, 0, stream>>>(Wp, wpt);

    // QKV: 3*4*16*64 = 12288 waves, 4 waves/block
    k_qkv<<<3072, 128, 0, stream>>>(xb, wqkvt, qb, kb, vtb);

    // fused scores+softmax: one block per (b,h,t-block) = 4096 blocks of 8 waves
    // dynamic LDS: staged K slab (TT*KROW bf16) + 2*128 floats of reductions
    const size_t shbytes = (size_t)TT * KROW * 2 + 256 * sizeof(float);
    k_attn<<<4096, 256, shbytes, stream>>>(qb, kb, mask, causal, att);

    // att @ v: 4096 waves
    k_av<<<512, 256, 0, stream>>>(att, vtb, causal, houtb);

    // projection: 4096 waves
    k_proj<<<512, 256, 0, stream>>>(houtb, wpt, bproj, out);
}
